// HybridKernelModel_65481071402920
// MI455X (gfx1250) — compile-verified, matrix operand order checked by reading, not run on hardware
//
#include <hip/hip_runtime.h>

#define NROWS  4096
#define KDIM   512
#define BLK    128
#define KT     32
#define LDP    40              // padded LDS row stride in bf16 elems (80 B, 16B-aligned)
#define KSTEPS (KDIM / KT)     // 16
#define NBUF   3               // async pipeline depth (2 stages in flight)
#define USE_ASYNC_LDS 1

typedef __attribute__((ext_vector_type(16))) __bf16        v16bf;
typedef __attribute__((ext_vector_type(8)))  float         v8f;
typedef __attribute__((ext_vector_type(4)))  unsigned int  v4u;
typedef unsigned short ushort_t;

union Frag {
  v4u   u[2];   // 32 bytes
  v16bf bf;     // 16 x bf16
};

// Split 8 fp32 into truncated-bf16 hi + bf16 lo residual, packed with v_perm_b32.
// hi = trunc16(f); lo = bf16(f - hi). (lo absorbs truncation err; dropped lo*lo ~ 2^-16 rel.)
__device__ __forceinline__ void split8(const float4 f0, const float4 f1,
                                       v4u& hu, v4u& lu) {
  const float f[8] = {f0.x, f0.y, f0.z, f0.w, f1.x, f1.y, f1.z, f1.w};
#pragma unroll
  for (int i = 0; i < 4; i++) {
    const unsigned ua = __float_as_uint(f[2 * i + 0]);
    const unsigned ub = __float_as_uint(f[2 * i + 1]);
    hu[i] = __builtin_amdgcn_perm(ub, ua, 0x07060302u);    // {b.hi16, a.hi16}
    const float ra = f[2 * i + 0] - __uint_as_float(ua & 0xFFFF0000u);
    const float rb = f[2 * i + 1] - __uint_as_float(ub & 0xFFFF0000u);
    lu[i] = __builtin_amdgcn_perm(__float_as_uint(rb), __float_as_uint(ra), 0x07060302u);
  }
}

// ---- CDNA5 async global->LDS staging (ASYNCcnt path, saddr form) -----------
__device__ __forceinline__ void async_copy16_s(void* lds, const void* sbase,
                                               unsigned voff) {
#if USE_ASYNC_LDS
  const unsigned loff = (unsigned)(unsigned long long)lds;   // low 32b = LDS offset
  asm volatile("global_load_async_to_lds_b128 %0, %1, %2"
               :: "v"(loff), "v"(voff), "s"(sbase) : "memory");
#else
  *(v4u*)lds = *(const v4u*)((const char*)sbase + voff);
#endif
}

template <int N>
__device__ __forceinline__ void async_wait_le() {
#if USE_ASYNC_LDS
  asm volatile("s_wait_asynccnt %0" :: "i"(N) : "memory");
#endif
}

// ---- shared WMMA tile compute: 8 C-tiles x 3 split products ----------------
__device__ __forceinline__ void wmma_tiles(
    const ushort_t (* __restrict__ sAhi)[LDP], const ushort_t (* __restrict__ sAlo)[LDP],
    const ushort_t (* __restrict__ sBhi)[LDP], const ushort_t (* __restrict__ sBlo)[LDP],
    int wm, int wn, int lm, int r0, int r1, v8f acc[4][2]) {
  Frag bH[2], bL[2];
#pragma unroll
  for (int nt = 0; nt < 2; nt++) {
    const int n = wn * 32 + nt * 16 + lm;
    bH[nt].u[0] = *(const v4u*)&sBhi[n][r0];
    bH[nt].u[1] = *(const v4u*)&sBhi[n][r1];
    bL[nt].u[0] = *(const v4u*)&sBlo[n][r0];
    bL[nt].u[1] = *(const v4u*)&sBlo[n][r1];
  }
#pragma unroll
  for (int mt = 0; mt < 4; mt++) {
    const int m = wm * 64 + mt * 16 + lm;
    Frag aH, aL;
    aH.u[0] = *(const v4u*)&sAhi[m][r0];
    aH.u[1] = *(const v4u*)&sAhi[m][r1];
    aL.u[0] = *(const v4u*)&sAlo[m][r0];
    aL.u[1] = *(const v4u*)&sAlo[m][r1];
#pragma unroll
    for (int nt = 0; nt < 2; nt++) {
      acc[mt][nt] = __builtin_amdgcn_wmma_f32_16x16x32_bf16(
          false, aH.bf, false, bH[nt].bf, (short)0, acc[mt][nt], false, false);
      acc[mt][nt] = __builtin_amdgcn_wmma_f32_16x16x32_bf16(
          false, aH.bf, false, bL[nt].bf, (short)0, acc[mt][nt], false, false);
      acc[mt][nt] = __builtin_amdgcn_wmma_f32_16x16x32_bf16(
          false, aL.bf, false, bH[nt].bf, (short)0, acc[mt][nt], false, false);
    }
  }
}

// ---- fused epilogue: Out = w * exp(-max(a2 + b2 - 2*dot, 0)) ---------------
__device__ __forceinline__ void epilogue(const v8f acc[4][2],
                                         const float* __restrict__ A2,
                                         const float* __restrict__ B2,
                                         const float* __restrict__ Wp,
                                         float* __restrict__ Out,
                                         int blockM, int blockN,
                                         int wm, int wn, int lm, int hf) {
  const float wv = Wp[0];
#pragma unroll
  for (int mt = 0; mt < 4; mt++) {
#pragma unroll
    for (int nt = 0; nt < 2; nt++) {
      const int col = blockN + wn * 32 + nt * 16 + lm;
      const float b2v = B2[col];
#pragma unroll
      for (int r = 0; r < 8; r++) {
        const int row = blockM + wm * 64 + mt * 16 + (hf ? 8 : 0) + r;
        const float d2 = fmaxf(A2[row] + b2v - 2.0f * acc[mt][nt][r], 0.0f);
        Out[(size_t)row * NROWS + col] = wv * __expf(-d2);
      }
    }
  }
}

// ---------------- row-norm pre-kernel ---------------------------------------
__global__ void row_norms(const float* __restrict__ X, float* __restrict__ O) {
  const int row = blockIdx.x;
  const int t   = threadIdx.x;                 // 128 threads = 4 waves
  const float* p = X + (size_t)row * KDIM;
  float s = 0.f;
#pragma unroll
  for (int c = t; c < KDIM; c += 128) {
    const float v = p[c];
    s = fmaf(v, v, s);
  }
#pragma unroll
  for (int off = 16; off > 0; off >>= 1) s += __shfl_xor(s, off, 32);
  __shared__ float ps[4];
  if ((t & 31) == 0) ps[t >> 5] = s;
  __syncthreads();
  if (t == 0) O[row] = ps[0] + ps[1] + ps[2] + ps[3];
}

// ---------------- one-shot fp32 -> (bf16 hi, bf16 lo) split -----------------
__global__ __launch_bounds__(256)
void split_bf16(const float* __restrict__ X, ushort_t* __restrict__ Hi,
                ushort_t* __restrict__ Lo) {
  const size_t idx = ((size_t)blockIdx.x * 256 + threadIdx.x) * 8;
  const float4 f0 = *(const float4*)(X + idx);
  const float4 f1 = *(const float4*)(X + idx + 4);
  v4u hu, lu;
  split8(f0, f1, hu, lu);
  *(v4u*)(Hi + idx) = hu;
  *(v4u*)(Lo + idx) = lu;
}

// ---------------- fast GEMM: pre-split bf16 + 3-deep async pipeline ---------
__global__ __launch_bounds__(256)
void rbf_gram_fast(const ushort_t* __restrict__ Ahi, const ushort_t* __restrict__ Alo,
                   const ushort_t* __restrict__ Bhi, const ushort_t* __restrict__ Blo,
                   const float* __restrict__ A2, const float* __restrict__ B2,
                   const float* __restrict__ Wp, float* __restrict__ Out) {
  __shared__ __align__(16) ushort_t sAhi[NBUF][BLK][LDP];
  __shared__ __align__(16) ushort_t sAlo[NBUF][BLK][LDP];
  __shared__ __align__(16) ushort_t sBhi[NBUF][BLK][LDP];
  __shared__ __align__(16) ushort_t sBlo[NBUF][BLK][LDP];

  const int tid  = threadIdx.x;
  const int wave = tid >> 5, lane = tid & 31;
  const int wm = wave >> 2, wn = wave & 3;
  const int hf = lane >> 4, lm = lane & 15;
  const int blockM = blockIdx.x * BLK, blockN = blockIdx.y * BLK;
  const int r0 = hf ? 8 : 0, r1 = hf ? 24 : 16;   // ISA 7.12.2 16-bit A lane layout

  const v8f vzero = {0.f, 0.f, 0.f, 0.f, 0.f, 0.f, 0.f, 0.f};
  v8f acc[4][2];
#pragma unroll
  for (int mt = 0; mt < 4; mt++)
#pragma unroll
    for (int nt = 0; nt < 2; nt++) acc[mt][nt] = vzero;

  // Per-thread chunk geometry: 512 x 16B chunks per array per k-step, 2/thread.
  // Byte voff into a [*, KDIM] bf16 panel is identical for all 4 arrays and
  // invariant across k-steps (k advance lives in the uniform SGPR base).
  const int row0 = tid >> 2;                       // 0..63
  const int seg0 = (tid & 3) * 8;                  // bf16 elems: 0,8,16,24
  const unsigned voff0 = (unsigned)(row0 * KDIM + seg0) * 2u;
  const unsigned voff1 = voff0 + 64u * KDIM * 2u;  // +64 rows

  const ushort_t* gA_hi = Ahi + (size_t)blockM * KDIM;
  const ushort_t* gA_lo = Alo + (size_t)blockM * KDIM;
  const ushort_t* gB_hi = Bhi + (size_t)blockN * KDIM;
  const ushort_t* gB_lo = Blo + (size_t)blockN * KDIM;

  auto stage = [&](int kt, int buf) {
    const ushort_t* pAh = gA_hi + kt * KT;    // uniform SGPR base advance
    const ushort_t* pAl = gA_lo + kt * KT;
    const ushort_t* pBh = gB_hi + kt * KT;
    const ushort_t* pBl = gB_lo + kt * KT;
    async_copy16_s(&sAhi[buf][row0][seg0], pAh, voff0);
    async_copy16_s(&sAlo[buf][row0][seg0], pAl, voff0);
    async_copy16_s(&sBhi[buf][row0][seg0], pBh, voff0);
    async_copy16_s(&sBlo[buf][row0][seg0], pBl, voff0);
    async_copy16_s(&sAhi[buf][row0 + 64][seg0], pAh, voff1);
    async_copy16_s(&sAlo[buf][row0 + 64][seg0], pAl, voff1);
    async_copy16_s(&sBhi[buf][row0 + 64][seg0], pBh, voff1);
    async_copy16_s(&sBlo[buf][row0 + 64][seg0], pBl, voff1);
  };

  // Prologue: two stages in flight (8 async ops each, done in issue order).
  stage(0, 0);
  stage(1, 1);

  for (int kt = 0; kt < KSTEPS; kt++) {
    // <=8 in flight  ==>  stage kt has fully landed in LDS (in-order done).
    if (kt + 1 < KSTEPS) async_wait_le<8>();
    else                 async_wait_le<0>();
    __syncthreads();                 // all waves' stage-kt data visible;
                                     // all reads of buf[(kt+2)%3] retired
    if (kt + 2 < KSTEPS) stage(kt + 2, (kt + 2) % NBUF);
    const int cur = kt % NBUF;
    wmma_tiles(sAhi[cur], sAlo[cur], sBhi[cur], sBlo[cur], wm, wn, lm, r0, r1, acc);
  }

  epilogue(acc, A2, B2, Wp, Out, blockM, blockN, wm, wn, lm, hf);
}

// ---------------- fallback GEMM: in-kernel fp32 split (small d_ws) ----------
__global__ __launch_bounds__(256)
void rbf_gram_conv(const float* __restrict__ A, const float* __restrict__ B,
                   const float* __restrict__ A2, const float* __restrict__ B2,
                   const float* __restrict__ Wp, float* __restrict__ Out) {
  __shared__ __align__(16) ushort_t sAhi[2][BLK][LDP];
  __shared__ __align__(16) ushort_t sAlo[2][BLK][LDP];
  __shared__ __align__(16) ushort_t sBhi[2][BLK][LDP];
  __shared__ __align__(16) ushort_t sBlo[2][BLK][LDP];

  const int tid  = threadIdx.x;
  const int wave = tid >> 5, lane = tid & 31;
  const int wm = wave >> 2, wn = wave & 3;
  const int hf = lane >> 4, lm = lane & 15;
  const int blockM = blockIdx.x * BLK, blockN = blockIdx.y * BLK;
  const int r0 = hf ? 8 : 0, r1 = hf ? 24 : 16;
  const int lrow = tid >> 2;          // 0..63
  const int lcol = (tid & 3) * 8;     // 0,8,16,24

  const v8f vzero = {0.f, 0.f, 0.f, 0.f, 0.f, 0.f, 0.f, 0.f};
  v8f acc[4][2];
#pragma unroll
  for (int mt = 0; mt < 4; mt++)
#pragma unroll
    for (int nt = 0; nt < 2; nt++) acc[mt][nt] = vzero;

  float4 pa[2][2], pb[2][2];

  auto gload = [&](int kt) {
    const float* abase = A + (size_t)(blockM + lrow) * KDIM + kt * KT + lcol;
    pa[0][0] = *(const float4*)(abase);
    pa[0][1] = *(const float4*)(abase + 4);
    pa[1][0] = *(const float4*)(abase + (size_t)64 * KDIM);
    pa[1][1] = *(const float4*)(abase + (size_t)64 * KDIM + 4);
    const float* bbase = B + (size_t)(blockN + lrow) * KDIM + kt * KT + lcol;
    pb[0][0] = *(const float4*)(bbase);
    pb[0][1] = *(const float4*)(bbase + 4);
    pb[1][0] = *(const float4*)(bbase + (size_t)64 * KDIM);
    pb[1][1] = *(const float4*)(bbase + (size_t)64 * KDIM + 4);
  };

  auto lstore = [&](int buf) {
    v4u hu, lu;
    split8(pa[0][0], pa[0][1], hu, lu);
    *(v4u*)&sAhi[buf][lrow][lcol] = hu;      *(v4u*)&sAlo[buf][lrow][lcol] = lu;
    split8(pa[1][0], pa[1][1], hu, lu);
    *(v4u*)&sAhi[buf][lrow + 64][lcol] = hu; *(v4u*)&sAlo[buf][lrow + 64][lcol] = lu;
    split8(pb[0][0], pb[0][1], hu, lu);
    *(v4u*)&sBhi[buf][lrow][lcol] = hu;      *(v4u*)&sBlo[buf][lrow][lcol] = lu;
    split8(pb[1][0], pb[1][1], hu, lu);
    *(v4u*)&sBhi[buf][lrow + 64][lcol] = hu; *(v4u*)&sBlo[buf][lrow + 64][lcol] = lu;
  };

  gload(0);
  lstore(0);
  __syncthreads();

  for (int kt = 0; kt < KSTEPS; kt++) {
    const int cur = kt & 1;
    if (kt + 1 < KSTEPS) gload(kt + 1);
    wmma_tiles(sAhi[cur], sAlo[cur], sBhi[cur], sBlo[cur], wm, wn, lm, r0, r1, acc);
    if (kt + 1 < KSTEPS) {
      lstore(1 - cur);
      __syncthreads();
    }
  }

  epilogue(acc, A2, B2, Wp, Out, blockM, blockN, wm, wn, lm, hf);
}

extern "C" void kernel_launch(void* const* d_in, const int* in_sizes, int n_in,
                              void* d_out, int out_size, void* d_ws, size_t ws_size,
                              hipStream_t stream) {
  const float* a = (const float*)d_in[0];   // [4096, 512] f32
  const float* b = (const float*)d_in[1];   // [4096, 512] f32
  const float* w = (const float*)d_in[2];   // [1] f32
  float* out = (float*)d_out;               // [4096, 4096] f32

  const size_t elems = (size_t)NROWS * KDIM;
  float* a2 = (float*)d_ws;
  float* b2 = a2 + NROWS;
  ushort_t* Ahi = (ushort_t*)(b2 + NROWS);
  ushort_t* Alo = Ahi + elems;
  ushort_t* Bhi = Alo + elems;
  ushort_t* Blo = Bhi + elems;
  const size_t need = 2 * NROWS * sizeof(float) + 4 * elems * sizeof(ushort_t);

  row_norms<<<NROWS, 128, 0, stream>>>(a, a2);
  row_norms<<<NROWS, 128, 0, stream>>>(b, b2);

  dim3 grid(NROWS / BLK, NROWS / BLK);      // 32 x 32
  if (ws_size >= need) {
    const int sblocks = (int)(elems / (8 * 256));   // 1024
    split_bf16<<<sblocks, 256, 0, stream>>>(a, Ahi, Alo);
    split_bf16<<<sblocks, 256, 0, stream>>>(b, Bhi, Blo);
    rbf_gram_fast<<<grid, 256, 0, stream>>>(Ahi, Alo, Bhi, Blo, a2, b2, w, out);
  } else {
    rbf_gram_conv<<<grid, 256, 0, stream>>>(a, b, a2, b2, w, out);
  }
}
